// TopologyPredictor_22265110463270
// MI455X (gfx1250) — matrix operand order repaired and checked by reading, not compiled, not run on hardware
//
#include <hip/hip_runtime.h>
#include <math.h>

typedef _Float16 f16;
typedef __attribute__((ext_vector_type(16))) _Float16 v16h;
typedef __attribute__((ext_vector_type(8)))  _Float16 v8h;
typedef __attribute__((ext_vector_type(8)))  float    v8f;

#define BB 2
#define SS 4096
#define DD 512
#define D2 256
#define KTOP 32
#define NEG_INF (-__builtin_inff())

// ---------- WMMA operand loads (wave32, v_wmma_f32_16x16x32_f16) ----------
// A (16x32, 16-bit): lane L holds row M=L&15; one contiguous 16-elem chunk
// at K = (L>>4)*16 (same k order as B).
__device__ __forceinline__ v16h load_a(const f16* base, int ld, int row0, int k0, int lane) {
  const f16* p = base + (size_t)(row0 + (lane & 15)) * ld + k0 + ((lane >> 4) << 4);
  return *(const v16h*)p;
}
// B (32x16, 16-bit): lane L holds column N=L&15; one contiguous 16-elem chunk
// at K = (L>>4)*16. Source is stored transposed (column n == row n of source).
__device__ __forceinline__ v16h load_b(const f16* baseT, int ld, int col0, int k0, int lane) {
  const f16* p = baseT + (size_t)(col0 + (lane & 15)) * ld + k0 + ((lane >> 4) << 4);
  return *(const v16h*)p;
}

__device__ __forceinline__ v8f wmma_f16(v16h a, v16h b, v8f c) {
  return __builtin_amdgcn_wmma_f32_16x16x32_f16(false, a, false, b, (short)0, c, false, false);
}

// ---------- prep kernels ----------
__global__ void f32_to_f16_kernel(const float* __restrict__ in, f16* __restrict__ out, int n) {
  for (int i = blockIdx.x * blockDim.x + threadIdx.x; i < n; i += gridDim.x * blockDim.x)
    out[i] = (f16)in[i];
}

// W is KxN row-major f32; Wt is NxK row-major f16 (Wt[n*K+k] = W[k*N+n])
__global__ void transpose_to_f16_kernel(const float* __restrict__ W, f16* __restrict__ Wt, int K, int N) {
  int i = blockIdx.x * blockDim.x + threadIdx.x;
  if (i < K * N) {
    int n = i / K, k = i % K;
    Wt[i] = (f16)W[(size_t)k * N + n];
  }
}

// ---------- WMMA GEMM: out(f16, MxN) = [gelu](A @ W + bias), W pre-transposed ----------
// One wave -> one 16x64 output strip (4 accumulators share each A operand: 4x A reuse).
__global__ void gemm_wmma_f16(const f16* __restrict__ A, const f16* __restrict__ Wt,
                              const float* __restrict__ bias, f16* __restrict__ out,
                              int M, int K, int N, int apply_gelu) {
  int wave = (blockIdx.x * blockDim.x + threadIdx.x) >> 5;
  int lane = threadIdx.x & 31;
  int groupsN = N >> 6;                      // 64-wide column groups
  int total = (M >> 4) * groupsN;
  if (wave >= total) return;                 // uniform per wave -> EXEC all-1s for WMMA
  int tm = (wave / groupsN) << 4;
  int tn = (wave % groupsN) << 6;

  v8f acc[4] = {v8f{}, v8f{}, v8f{}, v8f{}};
  for (int k0 = 0; k0 < K; k0 += 32) {
    v16h a = load_a(A, K, tm, k0, lane);
#pragma unroll
    for (int t = 0; t < 4; ++t) {
      v16h b = load_b(Wt, K, tn + (t << 4), k0, lane);
      acc[t] = wmma_f16(a, b, acc[t]);
    }
  }

  int rbase = tm + ((lane >> 4) << 3);
#pragma unroll
  for (int t = 0; t < 4; ++t) {
    int col = tn + (t << 4) + (lane & 15);
    float bc = bias[col];
#pragma unroll
    for (int v = 0; v < 8; ++v) {
      float s = acc[t][v] + bc;
      if (apply_gelu) s = 0.5f * s * (1.0f + erff(s * 0.70710678118654752f)); // exact GELU
      out[(size_t)(rbase + v) * N + col] = (f16)s;
    }
  }
}

// ---------- fused scores + loc_bias + causal mask + streaming top-32 + softmax ----------
// One wave (block of 32) per (batch, 16-row query tile). The query tile's A operands
// (16 k-steps x v16h = 128 VGPRs) are register-resident across the whole causal loop;
// scores never touch HBM.
__global__ void scores_topk_kernel(const f16* __restrict__ intents, const f16* __restrict__ features,
                                   const float* __restrict__ loc_bias,
                                   int* __restrict__ out_idx, float* __restrict__ out_w) {
  __shared__ float tile[256];        // 16x16 score tile
  __shared__ float hs[16 * KTOP];    // per-row top-32 scores
  __shared__ int   hi[16 * KTOP];    // per-row top-32 indices

  int lane  = threadIdx.x;
  int bid   = blockIdx.x;
  int batch = bid / (SS >> 4);
  int ti    = bid % (SS >> 4);
  int q0    = ti << 4;

  const f16* I = intents  + (size_t)batch * SS * DD;
  const f16* F = features + (size_t)batch * SS * DD;

  // Pin the query (A) tile in registers for the entire causal sweep.
  v16h areg[DD / 32];
#pragma unroll
  for (int k = 0; k < DD / 32; ++k) areg[k] = load_a(I, DD, q0, k << 5, lane);

  float minv = NEG_INF;
  int   minp = 0;
  if (lane < 16) {
    for (int t = 0; t < KTOP; ++t) { hs[lane * KTOP + t] = NEG_INF; hi[lane * KTOP + t] = 0; }
  }
  __syncthreads();

  const float scale = 0.04419417382415922f;  // 1/sqrt(512)
  int col  = lane & 15;
  int rsel = (lane >> 4) << 3;

  for (int jt = 0; jt <= ti; ++jt) {         // causal: only key tiles j <= i
    v8f acc = {};
#pragma unroll
    for (int k = 0; k < DD / 32; ++k) {
      v16h b = load_b(F, DD, jt << 4, k << 5, lane);
      acc = wmma_f16(areg[k], b, acc);
    }
    if (jt < ti)  // prefetch next key tile (global_prefetch_b8)
      __builtin_prefetch((const void*)(F + (size_t)(((jt + 1) << 4) + (lane & 15)) * DD + ((lane >> 4) << 8)), 0, 1);

    int kj = (jt << 4) + col;
#pragma unroll
    for (int v = 0; v < 8; ++v) {
      int qi  = q0 + rsel + v;
      int rel = kj - qi;
      int bidx = rel; if (bidx < -64) bidx = -64; if (bidx > 63) bidx = 63; bidx += 64;
      float s = acc[v] * scale + loc_bias[bidx];
      if (rel > 0) s = NEG_INF;              // causal mask (padding mask is all-true)
      tile[(rsel + v) * 16 + col] = s;
    }
    __syncthreads();

    if (lane < 16) {                          // lane owns query row q0+lane
      float* hrow  = hs + lane * KTOP;
      int*   hirow = hi + lane * KTOP;
      const float* trow = tile + lane * 16;
      for (int n = 0; n < 16; ++n) {
        float s = trow[n];
        if (s > minv) {                       // replace current min, rescan
          hrow[minp] = s; hirow[minp] = (jt << 4) + n;
          minv = hrow[0]; minp = 0;
          for (int t = 1; t < KTOP; ++t) { float h = hrow[t]; if (h < minv) { minv = h; minp = t; } }
        }
      }
    }
    __syncthreads();
  }

  if (lane < 16) {
    float* hrow  = hs + lane * KTOP;
    int*   hirow = hi + lane * KTOP;
    // selection sort descending (32 entries)
    for (int a2 = 0; a2 < KTOP - 1; ++a2) {
      int best = a2; float bv = hrow[a2];
      for (int t = a2 + 1; t < KTOP; ++t) if (hrow[t] > bv) { bv = hrow[t]; best = t; }
      if (best != a2) {
        float ts = hrow[a2]; hrow[a2] = hrow[best]; hrow[best] = ts;
        int   tx = hirow[a2]; hirow[a2] = hirow[best]; hirow[best] = tx;
      }
    }
    float m = hrow[0];
    float sum = 0.0f;
    for (int t = 0; t < KTOP; ++t) sum += expf(hrow[t] - m);
    float inv = 1.0f / sum;
    size_t obase = ((size_t)batch * SS + q0 + lane) * KTOP;
    for (int t = 0; t < KTOP; ++t) {
      out_idx[obase + t] = hirow[t];
      out_w[obase + t]   = expf(hrow[t] - m) * inv;
    }
  }
}

extern "C" void kernel_launch(void* const* d_in, const int* in_sizes, int n_in,
                              void* d_out, int out_size, void* d_ws, size_t ws_size,
                              hipStream_t stream) {
  (void)in_sizes; (void)n_in; (void)out_size; (void)ws_size;
  const float* x        = (const float*)d_in[0];
  const float* Wi1      = (const float*)d_in[1];
  const float* bi1      = (const float*)d_in[2];
  const float* Wi2      = (const float*)d_in[3];
  const float* bi2      = (const float*)d_in[4];
  const float* Wf1      = (const float*)d_in[5];
  const float* bf1      = (const float*)d_in[6];
  const float* Wf2      = (const float*)d_in[7];
  const float* bf2      = (const float*)d_in[8];
  const float* loc_bias = (const float*)d_in[9];
  // d_in[10..13] = adaptive-k predictor (unused for outputs); d_in[14] = mask (all true)

  char* ws = (char*)d_ws;
  size_t off = 0;
  auto wsalloc = [&](size_t bytes) -> void* {
    void* p = ws + off; off += (bytes + 255) & ~(size_t)255; return p;
  };
  f16* x16    = (f16*)wsalloc((size_t)BB * SS * DD * sizeof(f16));
  f16* h16    = (f16*)wsalloc((size_t)BB * SS * D2 * sizeof(f16));  // reused for both MLP hiddens
  f16* ints16 = (f16*)wsalloc((size_t)BB * SS * DD * sizeof(f16));
  f16* fts16  = (f16*)wsalloc((size_t)BB * SS * DD * sizeof(f16));
  f16* Wi1t   = (f16*)wsalloc((size_t)DD * D2 * sizeof(f16));
  f16* Wi2t   = (f16*)wsalloc((size_t)D2 * DD * sizeof(f16));
  f16* Wf1t   = (f16*)wsalloc((size_t)DD * D2 * sizeof(f16));
  f16* Wf2t   = (f16*)wsalloc((size_t)D2 * DD * sizeof(f16));

  int nx = BB * SS * DD;
  f32_to_f16_kernel<<<(nx + 255) / 256, 256, 0, stream>>>(x, x16, nx);
  int nw = DD * D2;
  transpose_to_f16_kernel<<<(nw + 255) / 256, 256, 0, stream>>>(Wi1, Wi1t, DD, D2);
  transpose_to_f16_kernel<<<(nw + 255) / 256, 256, 0, stream>>>(Wi2, Wi2t, D2, DD);
  transpose_to_f16_kernel<<<(nw + 255) / 256, 256, 0, stream>>>(Wf1, Wf1t, DD, D2);
  transpose_to_f16_kernel<<<(nw + 255) / 256, 256, 0, stream>>>(Wf2, Wf2t, D2, DD);

  const int M = BB * SS;                 // 8192 rows
  {
    int waves = (M >> 4) * (D2 >> 6);    // 2048 strips of 16x64
    gemm_wmma_f16<<<waves / 4, 128, 0, stream>>>(x16, Wi1t, bi1, h16, M, DD, D2, 1);
  }
  {
    int waves = (M >> 4) * (DD >> 6);    // 4096 strips
    gemm_wmma_f16<<<waves / 4, 128, 0, stream>>>(h16, Wi2t, bi2, ints16, M, D2, DD, 0);
  }
  {
    int waves = (M >> 4) * (D2 >> 6);
    gemm_wmma_f16<<<waves / 4, 128, 0, stream>>>(x16, Wf1t, bf1, h16, M, DD, D2, 1);
  }
  {
    int waves = (M >> 4) * (DD >> 6);
    gemm_wmma_f16<<<waves / 4, 128, 0, stream>>>(h16, Wf2t, bf2, fts16, M, D2, DD, 0);
  }

  int*   out_idx = (int*)d_out;                              // top_indices (B,S,32) int32
  float* out_w   = (float*)d_out + (size_t)BB * SS * KTOP;   // weights     (B,S,32) f32
  scores_topk_kernel<<<BB * (SS >> 4), 32, 0, stream>>>(ints16, fts16, loc_bias, out_idx, out_w);
}
